// DeepTemplateMatchingModule_86870008529409
// MI455X (gfx1250) — matrix operand order repaired and checked by reading, not compiled, not run on hardware
//
#include <hip/hip_runtime.h>
#include <hip/hip_bf16.h>
#include <math.h>

typedef __attribute__((ext_vector_type(16))) _Float16 v16h;
typedef __attribute__((ext_vector_type(4)))  _Float16 v4h;
typedef __attribute__((ext_vector_type(8)))  float    v8f;

// ---------------------------------------------------------------------------
// CDNA5 wave32 WMMA 16x16x32 f16 fragment layouts as staged here (contiguous k order):
//  A 16x32: lane = m + 16*(k>>4),     half-slot i = k&15
//  B 32x16: lane = n + 16*(k>>4),     half-slot i = k&15
//  C 16x16 f32: vgpr r -> m = (lane>>4)*8 + r, n = lane&15
// We stage LDS tiles in *fragment order*: a 16x16x32 operand tile occupies
// 512 halves laid out [lane][16 halves], so a fragment load is one 32B
// contiguous LDS read (ds_load_b128 x2) per lane.
// ---------------------------------------------------------------------------

__device__ __forceinline__ float sigmoidf_(float x) { return 1.f / (1.f + expf(-x)); }

// ---------------------------------------------------------------------------
// Implicit-GEMM 5x5 VALID conv, f16 in/out, f32 accumulate via WMMA.
// Block: 128 threads (4 waves). C tile: COUT (M) x 64 (N = w positions).
// K = CIN*25 chunked by 32 (zero-padded weights), double-buffered LDS.
// ---------------------------------------------------------------------------
template <int COUT, int CIN>
__global__ void conv5x5_wmma_kernel(const _Float16* __restrict__ in,   // [CIN][Hin][Win]
                                    const _Float16* __restrict__ wp,   // [COUT][KPAD] prepacked
                                    const float*    __restrict__ bias, // [COUT]
                                    _Float16*       __restrict__ out,  // [COUT][Hout][Wout]
                                    int Hin, int Win) {
    constexpr int K       = CIN * 25;
    constexpr int KPAD    = (K + 31) & ~31;
    constexpr int WAVES_M = COUT / 16;      // 1, 2, 4
    constexpr int WAVES_N = 4 / WAVES_M;    // 4, 2, 1
    constexpr int NPW     = 64 / WAVES_N;   // n-width per wave
    constexpr int NFRAG   = NPW / 16;

    const int Hout = Hin - 4, Wout = Win - 4;
    const int h    = blockIdx.y;
    const int w0   = blockIdx.x * 64;
    const int tid  = threadIdx.x;
    const int lane = tid & 31, wid = tid >> 5;
    const int m0w  = (wid % WAVES_M) * 16;
    const int n0w  = (wid / WAVES_M) * NPW;

    __shared__ __align__(32) _Float16 As[2][WAVES_M * 512];
    __shared__ __align__(32) _Float16 Bs[2][4 * 512];

    // weights -> fragment-order LDS (vectorized, KPAD % 4 == 0)
    auto fillA = [&](int buf, int kc) {
#pragma unroll
        for (int e = tid; e < COUT * 8; e += 128) {
            const int m = e >> 3, kq = e & 7;
            const v4h v = *(const v4h*)(wp + m * KPAD + kc + kq * 4);
            const int hi = kq >> 2;
            const int i0 = (kq & 3) << 2;
            *(v4h*)(&As[buf][(m >> 4) * 512 + ((m & 15) + (hi << 4)) * 16 + i0]) = v;
        }
    };
    // im2col tile -> fragment-order LDS (scalar gathers; coalesced over n)
    auto fillB = [&](int buf, int kc) {
        for (int e = tid; e < 2048; e += 128) {
            const int kk = e >> 6, n = e & 63;
            const int k  = kc + kk;
            _Float16 v   = (_Float16)0.f;
            if (K == KPAD || k < K) {
                const int c  = k / 25;
                const int r  = k - c * 25;
                const int kh = r / 5;
                const int kw = r - kh * 5;
                const int iw = w0 + n + kw;
                if (iw < Win)
                    v = in[((long)c * Hin + (h + kh)) * Win + iw];
            }
            Bs[buf][(n >> 4) * 512 + ((n & 15) + ((kk >> 4) << 4)) * 16 + (kk & 15)] = v;
        }
    };

    v8f zero = {};
    v8f acc[NFRAG];
#pragma unroll
    for (int f = 0; f < NFRAG; ++f) acc[f] = zero;

    fillA(0, 0);
    fillB(0, 0);
    __syncthreads();
    int buf = 0;
    for (int kc = 0; kc < KPAD; kc += 32) {
        const int nxt = buf ^ 1;
        if (kc + 32 < KPAD) { fillA(nxt, kc + 32); fillB(nxt, kc + 32); }
        const v16h a = *(const v16h*)(&As[buf][(m0w >> 4) * 512 + lane * 16]);
#pragma unroll
        for (int f = 0; f < NFRAG; ++f) {
            const v16h b = *(const v16h*)(&Bs[buf][((n0w >> 4) + f) * 512 + lane * 16]);
            acc[f] = __builtin_amdgcn_wmma_f32_16x16x32_f16(
                false, a, false, b, (short)0, acc[f], false, false);
        }
        __syncthreads();
        buf = nxt;
    }

    const int hi = lane >> 4, nl = lane & 15;
    const int HW = Hout * Wout;
    float bm[8];
#pragma unroll
    for (int r = 0; r < 8; ++r) bm[r] = bias[m0w + hi * 8 + r];
#pragma unroll
    for (int f = 0; f < NFRAG; ++f) {
        const int w = w0 + n0w + f * 16 + nl;
        if (w < Wout) {
            const int idx = (m0w + hi * 8) * HW + h * Wout + w;
#pragma unroll
            for (int r = 0; r < 8; ++r)
                out[idx + r * HW] = (_Float16)(acc[f][r] + bm[r]);
        }
    }
}

// ---------------------------------------------------------------------------
// Generic LDS-tiled WMMA GEMM (no tails: grid covers M,N exactly, M%64==0,
// N%64==0, K%32==0 for every call in this module).
//   C[m,n] = sum_k A[m,k] * (B_NK ? B[n,k] : B[k,n]) + bias[n]
// Block: 128 threads / 4 waves, 64x64 C tile, double-buffered 32-deep chunks.
// blockIdx.z batches via element strides sA/sB/sC.
// ---------------------------------------------------------------------------
template <bool B_NK>
__global__ void gemm_wmma_kernel(const _Float16* __restrict__ A, int lda, long sA,
                                 const _Float16* __restrict__ B, int ldb, long sB,
                                 const float*    __restrict__ bias,
                                 float*          __restrict__ Cf,
                                 _Float16*       __restrict__ Ch,
                                 int ldc, long sC, int Kdim) {
    const _Float16* Ap = A + (long)blockIdx.z * sA + (long)(blockIdx.y * 64) * lda;
    const _Float16* Bp = B + (long)blockIdx.z * sB;
    const int n0  = blockIdx.x * 64;
    const int tid = threadIdx.x;
    const int lane = tid & 31, wid = tid >> 5;

    __shared__ __align__(32) _Float16 As[2][2048];
    __shared__ __align__(32) _Float16 Bs[2][2048];

    auto fillA = [&](int buf, int kc) {
#pragma unroll
        for (int e = tid; e < 512; e += 128) {
            const int m = e >> 3, kq = e & 7;
            const v4h v = *(const v4h*)(Ap + (long)m * lda + kc + kq * 4);
            const int hi = kq >> 2;
            const int i0 = (kq & 3) << 2;
            *(v4h*)(&As[buf][(m >> 4) * 512 + ((m & 15) + (hi << 4)) * 16 + i0]) = v;
        }
    };
    auto fillB = [&](int buf, int kc) {
        if (B_NK) {
#pragma unroll
            for (int e = tid; e < 512; e += 128) {
                const int n = e >> 3, kq = e & 7;
                const v4h v = *(const v4h*)(Bp + (long)(n0 + n) * ldb + kc + kq * 4);
                const int hi = kq >> 2;
                *(v4h*)(&Bs[buf][(n >> 4) * 512 + ((n & 15) + (hi << 4)) * 16 + (kq & 3) * 4]) = v;
            }
        } else {
#pragma unroll
            for (int e = tid; e < 512; e += 128) {
                const int k = e >> 4, nq = e & 15;
                const v4h v = *(const v4h*)(Bp + (long)(kc + k) * ldb + n0 + nq * 4);
#pragma unroll
                for (int j = 0; j < 4; ++j) {
                    const int n = nq * 4 + j;
                    Bs[buf][(n >> 4) * 512 + ((n & 15) + ((k >> 4) << 4)) * 16 + (k & 15)] = v[j];
                }
            }
        }
    };

    v8f zero = {};
    v8f acc[4];
#pragma unroll
    for (int f = 0; f < 4; ++f) acc[f] = zero;

    fillA(0, 0);
    fillB(0, 0);
    __syncthreads();
    int buf = 0;
    for (int kc = 0; kc < Kdim; kc += 32) {
        const int nxt = buf ^ 1;
        if (kc + 32 < Kdim) { fillA(nxt, kc + 32); fillB(nxt, kc + 32); }
        if (kc + 64 < Kdim)  // speculative prefetch of the chunk after next
            __builtin_prefetch(Ap + (long)(tid >> 1) * lda + kc + 64 + (tid & 1) * 16, 0, 1);
        const v16h a = *(const v16h*)(&As[buf][wid * 512 + lane * 16]);
#pragma unroll
        for (int f = 0; f < 4; ++f) {
            const v16h b = *(const v16h*)(&Bs[buf][f * 512 + lane * 16]);
            acc[f] = __builtin_amdgcn_wmma_f32_16x16x32_f16(
                false, a, false, b, (short)0, acc[f], false, false);
        }
        __syncthreads();
        buf = nxt;
    }

    const long crow = (long)blockIdx.z * sC +
                      (long)(blockIdx.y * 64 + wid * 16 + (lane >> 4) * 8) * ldc +
                      n0 + (lane & 15);
    float*    cf = Cf ? Cf + crow : nullptr;
    _Float16* ch = Ch ? Ch + crow : nullptr;
#pragma unroll
    for (int f = 0; f < 4; ++f) {
        const int n  = f * 16;
        const float bn = bias ? bias[n0 + n + (lane & 15)] : 0.f;
#pragma unroll
        for (int r = 0; r < 8; ++r) {
            const float v = acc[f][r] + bn;
            if (Cf) cf[r * ldc + n] = v;
            if (Ch) ch[r * ldc + n] = (_Float16)v;
        }
    }
}

// ---------------------------------------------------------------------------
// Small helper kernels
// ---------------------------------------------------------------------------
__global__ void f32_to_f16_kernel(const float* __restrict__ in, _Float16* __restrict__ out, long n) {
    for (long i = (long)blockIdx.x * 256 + threadIdx.x; i < n; i += (long)gridDim.x * 256)
        out[i] = (_Float16)in[i];
}

// pad each row of [rows][K] f32 to [rows][KPAD] f16 (zeros past K)
__global__ void pack_wrows_kernel(const float* __restrict__ in, _Float16* __restrict__ out,
                                  int rows, int K, int KPAD) {
    const long total = (long)rows * KPAD;
    for (long i = (long)blockIdx.x * 256 + threadIdx.x; i < total; i += (long)gridDim.x * 256) {
        const int m = (int)(i / KPAD), k = (int)(i % KPAD);
        out[i] = (k < K) ? (_Float16)in[(long)m * K + k] : (_Float16)0.f;
    }
}

__global__ void zero_h_kernel(float* __restrict__ h, _Float16* __restrict__ h16, int n) {
    const int i = blockIdx.x * 256 + threadIdx.x;
    if (i < n) { h[i] = 0.f; h16[i] = (_Float16)0.f; }
}

// maxpool window (2,5) stride (2,1); output written flat [c][ho][wo]
__global__ void maxpool25_kernel(const _Float16* __restrict__ in, _Float16* __restrict__ out,
                                 int C, int Hin, int Win, int Hout, int Wout) {
    const int total = C * Hout * Wout;
    const int i = blockIdx.x * 256 + threadIdx.x;
    if (i >= total) return;
    const int wo = i % Wout;
    const int t  = i / Wout;
    const int ho = t % Hout;
    const int c  = t / Hout;
    float m = -3.4e38f;
#pragma unroll
    for (int dh = 0; dh < 2; ++dh)
#pragma unroll
        for (int dw = 0; dw < 5; ++dw)
            m = fmaxf(m, (float)in[((long)c * Hin + 2 * ho + dh) * Win + wo + dw]);
    out[i] = (_Float16)m;
}

// GRU elementwise update: gates/hg are [1024][192] (r|z|n), h is [1024][64]
__global__ void gru_update_kernel(const float* __restrict__ gates, const float* __restrict__ hg,
                                  float* __restrict__ h, _Float16* __restrict__ h16,
                                  float* __restrict__ ys, _Float16* __restrict__ ys16) {
    const int i = blockIdx.x * 256 + threadIdx.x;
    if (i >= 1024 * 64) return;
    const int t = i >> 6, d = i & 63;
    const float* g  = gates + (long)t * 192;
    const float* hh = hg    + (long)t * 192;
    const float r  = sigmoidf_(g[d]       + hh[d]);
    const float z  = sigmoidf_(g[64 + d]  + hh[64 + d]);
    const float nn = tanhf(g[128 + d] + r * hh[128 + d]);
    const float hv = h[i];
    const float hn = (1.f - z) * nn + z * hv;
    h[i]    = hn;
    h16[i]  = (_Float16)hn;
    ys[i]   = hn;
    ys16[i] = (_Float16)hn;
}

template <typename OutT>
__global__ void softmax_rows_kernel(const float* __restrict__ in, OutT* __restrict__ out, int ncols) {
    const long row = blockIdx.x;
    in  += row * (long)ncols;
    out += row * (long)ncols;
    __shared__ float red[256];
    float mx = -3.4e38f;
    for (int i = threadIdx.x; i < ncols; i += 256) mx = fmaxf(mx, in[i]);
    red[threadIdx.x] = mx;
    __syncthreads();
    for (int s = 128; s > 0; s >>= 1) {
        if (threadIdx.x < s) red[threadIdx.x] = fmaxf(red[threadIdx.x], red[threadIdx.x + s]);
        __syncthreads();
    }
    mx = red[0];
    __syncthreads();
    float sum = 0.f;
    for (int i = threadIdx.x; i < ncols; i += 256) sum += expf(in[i] - mx);
    red[threadIdx.x] = sum;
    __syncthreads();
    for (int s = 128; s > 0; s >>= 1) {
        if (threadIdx.x < s) red[threadIdx.x] += red[threadIdx.x + s];
        __syncthreads();
    }
    const float inv = 1.f / red[0];
    for (int i = threadIdx.x; i < ncols; i += 256) out[i] = (OutT)(expf(in[i] - mx) * inv);
}

// per-batch n x n f16 transpose (blockIdx.z = batch)
__global__ void transpose_f16_kernel(const _Float16* __restrict__ in, _Float16* __restrict__ out, int n) {
    __shared__ _Float16 tile[32][33];
    const long base = (long)blockIdx.z * n * n;
    int x  = blockIdx.x * 32 + threadIdx.x;
    int y0 = blockIdx.y * 32 + threadIdx.y;
    for (int i = 0; i < 32; i += 8)
        tile[threadIdx.y + i][threadIdx.x] = in[base + (long)(y0 + i) * n + x];
    __syncthreads();
    x  = blockIdx.y * 32 + threadIdx.x;
    y0 = blockIdx.x * 32 + threadIdx.y;
    for (int i = 0; i < 32; i += 8)
        out[base + (long)(y0 + i) * n + x] = tile[threadIdx.x][threadIdx.y + i];
}

__global__ void attn_logits_kernel(const float* __restrict__ ev, const float* __restrict__ aw,
                                   const float* __restrict__ ab, float* __restrict__ logits) {
    const int i = blockIdx.x * 256 + threadIdx.x;
    if (i >= 8 * 1024) return;
    const float* v = ev + (long)i * 64;
    float s = ab[0];
#pragma unroll 8
    for (int d = 0; d < 64; ++d) s += v[d] * aw[d];
    logits[i] = s;
}

__global__ void pooled_kernel(const float* __restrict__ X, const float* __restrict__ ev,
                              const float* __restrict__ attn, float* __restrict__ pooled) {
    const int b = blockIdx.x, d = threadIdx.x;  // 64 threads
    float s = 0.f;
    for (int t = 0; t < 1024; ++t) {
        const long i = ((long)b * 1024 + t) * 64 + d;
        s += fabsf(X[i] - ev[i]) * attn[b * 1024 + t];
    }
    pooled[b * 64 + d] = s;
}

__global__ void head_kernel(const float* __restrict__ pooled,
                            const float* __restrict__ hw, const float* __restrict__ hb,
                            const float* __restrict__ cw, const float* __restrict__ cb,
                            float* __restrict__ out) {
    const int b = blockIdx.x, j = threadIdx.x;  // 128 threads
    __shared__ float hbuf[128];
    __shared__ float lg[2];
    float s = hb[j];
#pragma unroll 8
    for (int d = 0; d < 64; ++d) s += pooled[b * 64 + d] * hw[j * 64 + d];
    hbuf[j] = fmaxf(s, 0.f);
    __syncthreads();
    if (j < 2) {
        float l = cb[j];
        for (int k = 0; k < 128; ++k) l += hbuf[k] * cw[j * 128 + k];
        lg[j] = l;
    }
    __syncthreads();
    if (j < 2) {
        const float mx = fmaxf(lg[0], lg[1]);
        const float e0 = expf(lg[0] - mx), e1 = expf(lg[1] - mx);
        out[b * 2 + j] = (j == 0 ? e0 : e1) / (e0 + e1);
    }
}

// ---------------------------------------------------------------------------
// Orchestration
// ---------------------------------------------------------------------------
static inline int cdiv(int a, int b) { return (a + b - 1) / b; }

extern "C" void kernel_launch(void* const* d_in, const int* in_sizes, int n_in,
                              void* d_out, int out_size, void* d_ws, size_t ws_size,
                              hipStream_t stream) {
    (void)in_sizes; (void)n_in; (void)out_size; (void)ws_size;

    // geometry
    const int B = 8;
    const int H0 = 128, W0 = 1040;
    const int H1 = 124, W1 = 1036;
    const int H2 = 120, W2 = 1032;
    const int H3 = 116, W3 = 1028;
    const int HP = 58,  WP = 1024;
    const int FLAT = 3712, T = 1024, D = 64, G = 192;

    char* ws = (char*)d_ws;
    size_t off = 0;
    auto alloc = [&](size_t bytes) -> void* {
        off = (off + 255) & ~(size_t)255;
        void* p = ws + off;
        off += bytes;
        return p;
    };

    // ---- persistent buffers (both branches' GRU outputs)
    float*    ys32[2];
    _Float16* ys16[2];
    for (int br = 0; br < 2; ++br) {
        ys32[br] = (float*)alloc((size_t)B * T * D * 4);
        ys16[br] = (_Float16*)alloc((size_t)B * T * D * 2);
    }
    float* logits = (float*)alloc((size_t)B * T * 4);
    float* attnw  = (float*)alloc((size_t)B * T * 4);
    float* pooled = (float*)alloc((size_t)B * D * 4);

    const size_t mark = off;  // everything below is phase-scoped and reused

    // ---- branch-temp buffers
    _Float16* in16   = (_Float16*)alloc((size_t)B * H0 * W0 * 2);
    _Float16* wp1    = (_Float16*)alloc((size_t)16 * 32 * 2);
    _Float16* wp2    = (_Float16*)alloc((size_t)32 * 416 * 2);
    _Float16* wp3    = (_Float16*)alloc((size_t)64 * 800 * 2);
    _Float16* lin16  = (_Float16*)alloc((size_t)64 * FLAT * 2);
    _Float16* wih16  = (_Float16*)alloc((size_t)G * D * 2);
    _Float16* whh16  = (_Float16*)alloc((size_t)G * D * 2);
    _Float16* feat16 = (_Float16*)alloc((size_t)B * T * D * 2);
    float*    gates  = (float*)alloc((size_t)B * T * G * 4);
    float*    h32    = (float*)alloc((size_t)T * D * 4);
    _Float16* h16    = (_Float16*)alloc((size_t)T * D * 2);
    float*    hg     = (float*)alloc((size_t)T * G * 4);
    _Float16* act1   = (_Float16*)alloc((size_t)16 * H1 * W1 * 2);
    _Float16* act2   = (_Float16*)alloc((size_t)32 * H2 * W2 * 2);
    _Float16* act3   = (_Float16*)alloc((size_t)64 * H3 * W3 * 2);
    _Float16* mfeat  = (_Float16*)alloc((size_t)64 * HP * WP * 2);

    // ================= branch pipeline (br 0 = evaluation, 1 = template)
    for (int br = 0; br < 2; ++br) {
        const int base = 2 + br * 12;
        const float* x    = (const float*)d_in[br];  // (8,128,1040)
        const float* w1   = (const float*)d_in[base + 0];
        const float* b1   = (const float*)d_in[base + 1];
        const float* w2   = (const float*)d_in[base + 2];
        const float* b2   = (const float*)d_in[base + 3];
        const float* w3   = (const float*)d_in[base + 4];
        const float* b3   = (const float*)d_in[base + 5];
        const float* linw = (const float*)d_in[base + 6];
        const float* linb = (const float*)d_in[base + 7];
        const float* wih  = (const float*)d_in[base + 8];
        const float* whh  = (const float*)d_in[base + 9];
        const float* bih  = (const float*)d_in[base + 10];
        const float* bhh  = (const float*)d_in[base + 11];

        // ---- pack input + weights to f16 (conv weights K-padded)
        const long nin = (long)B * H0 * W0;
        f32_to_f16_kernel<<<2048, 256, 0, stream>>>(x, in16, nin);
        pack_wrows_kernel<<<64, 256, 0, stream>>>(w1, wp1, 16, 25, 32);
        pack_wrows_kernel<<<64, 256, 0, stream>>>(w2, wp2, 32, 400, 416);
        pack_wrows_kernel<<<256, 256, 0, stream>>>(w3, wp3, 64, 800, 800);
        pack_wrows_kernel<<<1024, 256, 0, stream>>>(linw, lin16, 64, FLAT, FLAT);
        pack_wrows_kernel<<<64, 256, 0, stream>>>(wih, wih16, G, D, D);
        pack_wrows_kernel<<<64, 256, 0, stream>>>(whh, whh16, G, D, D);

        // ---- per-batch conv stack -> maxpool -> linear
        for (int b = 0; b < B; ++b) {
            const _Float16* xin = in16 + (long)b * H0 * W0;
            conv5x5_wmma_kernel<16, 1><<<dim3(cdiv(W1, 64), H1, 1), 128, 0, stream>>>(
                xin, wp1, b1, act1, H0, W0);
            conv5x5_wmma_kernel<32, 16><<<dim3(cdiv(W2, 64), H2, 1), 128, 0, stream>>>(
                act1, wp2, b2, act2, H1, W1);
            conv5x5_wmma_kernel<64, 32><<<dim3(cdiv(W3, 64), H3, 1), 128, 0, stream>>>(
                act2, wp3, b3, act3, H2, W2);
            maxpool25_kernel<<<cdiv(64 * HP * WP, 256), 256, 0, stream>>>(
                act3, mfeat, 64, H3, W3, HP, WP);
            // feat[b] = mfeat(1024x3712) @ lin_w^T + lin_b   (f16 out)
            gemm_wmma_kernel<true><<<dim3(1, 16, 1), 128, 0, stream>>>(
                mfeat, FLAT, 0, lin16, FLAT, 0, linb,
                (float*)nullptr, feat16 + (long)b * T * D, D, 0, FLAT);
        }

        // ---- GRU input gates for all 8 steps: (8192 x 64) @ wih^T + bih
        gemm_wmma_kernel<true><<<dim3(3, 128, 1), 128, 0, stream>>>(
            feat16, D, 0, wih16, D, 0, bih,
            gates, (_Float16*)nullptr, G, 0, D);

        // ---- GRU recurrence over the 8 "steps" (scan over batch dim)
        zero_h_kernel<<<cdiv(T * D, 256), 256, 0, stream>>>(h32, h16, T * D);
        for (int s = 0; s < B; ++s) {
            gemm_wmma_kernel<true><<<dim3(3, 16, 1), 128, 0, stream>>>(
                h16, D, 0, whh16, D, 0, bhh,
                hg, (_Float16*)nullptr, G, 0, D);
            gru_update_kernel<<<cdiv(T * D, 256), 256, 0, stream>>>(
                gates + (long)s * T * G, hg, h32, h16,
                ys32[br] + (long)s * T * D, ys16[br] + (long)s * T * D);
        }
    }

    // ================= attention phase (overlays the branch-temp region)
    off = mark;
    float*    S    = (float*)alloc((size_t)B * T * T * 4);
    _Float16* P16  = (_Float16*)alloc((size_t)B * T * T * 2);
    _Float16* PT16 = (_Float16*)alloc((size_t)B * T * T * 2);
    float*    X    = (float*)alloc((size_t)B * T * D * 4);

    _Float16* evh = ys16[0];   // evaluation branch GRU output
    _Float16* tph = ys16[1];   // template branch GRU output
    float*    evf = ys32[0];

    // S[b,t,s] = sum_d tp[b,t,d] * ev[b,s,d]
    gemm_wmma_kernel<true><<<dim3(16, 16, B), 128, 0, stream>>>(
        tph, D, (long)T * D, evh, D, (long)T * D, (const float*)nullptr,
        S, (_Float16*)nullptr, T, (long)T * T, D);
    // row softmax over s -> P (f16)
    softmax_rows_kernel<_Float16><<<B * T, 256, 0, stream>>>(S, P16, T);
    // PT[b,s,t] = P[b,t,s]
    transpose_f16_kernel<<<dim3(32, 32, B), dim3(32, 8), 0, stream>>>(P16, PT16, T);
    // X[b,s,d] = sum_t P[b,t,s] * tp[b,t,d]  ==  PT @ tp
    gemm_wmma_kernel<false><<<dim3(1, 16, B), 128, 0, stream>>>(
        PT16, T, (long)T * T, tph, D, (long)T * D, (const float*)nullptr,
        X, (_Float16*)nullptr, D, (long)T * D, T);

    // attention pooling over ev
    const float* aw = (const float*)d_in[26];
    const float* ab = (const float*)d_in[27];
    attn_logits_kernel<<<cdiv(B * T, 256), 256, 0, stream>>>(evf, aw, ab, logits);
    softmax_rows_kernel<float><<<B, 256, 0, stream>>>(logits, attnw, T);
    pooled_kernel<<<B, 64, 0, stream>>>(X, evf, attnw, pooled);

    // classifier head
    head_kernel<<<B, 128, 0, stream>>>(
        pooled, (const float*)d_in[28], (const float*)d_in[29],
        (const float*)d_in[30], (const float*)d_in[31], (float*)d_out);
}
